// SkipAttention_343597384424
// MI455X (gfx1250) — compile-verified, matrix operand order checked by reading, not run on hardware
//
#include <hip/hip_runtime.h>
#include <stdint.h>

typedef __attribute__((ext_vector_type(16))) _Float16 v16h;
typedef __attribute__((ext_vector_type(8)))  float    v8f;

#define DIMX  512
#define HEADS 16
#define DH    32
#define NTOK  49
#define NPAD  64
#define NWIN  2048
// padded LDS row strides (in f16 elements); all keep 16B alignment for b128 loads
#define SPAD  520   // 1040B = 65*16B -> bank step 4: conflict-free 16-lane b128 reads
#define VPAD  72    // 144B = 9*16B
#define APAD  72
// element offsets of the LDS buffers within the single smem allocation
#define OFF_X   0
#define OFF_Q   (NPAD * SPAD)
#define OFF_K   (2 * NPAD * SPAD)
#define OFF_V   (3 * NPAD * SPAD)

union FragU { v16h h; uint32_t u[8]; uint4 q[2]; };

// A/B fragment loader for v_wmma_f32_16x16x32_f16.
// A: lane l -> row m0+l; K packed {0..7,16..23} (lanes 0-15) / {8..15,24..31} (lanes 16-31).
// Those K groups are two contiguous 16B chunks -> two b128 loads.
// B operand uses the identical layout over columns, so pass "B^T stored row-major".
__device__ __forceinline__ v16h load_frag(const _Float16* __restrict__ base,
                                          int row0, int k0, int stride, int lane) {
  const int half = lane >> 4;
  const int l    = lane & 15;
  const _Float16* p = base + (row0 + l) * stride + k0 + half * 8;
  FragU f;
  f.q[0] = *(const uint4*)(p);        // K = k0+half*8 .. +7
  f.q[1] = *(const uint4*)(p + 16);   // K = k0+16+half*8 .. +7
  return f.h;
}

__device__ __forceinline__ v8f zero8() {
  v8f a;
#pragma unroll
  for (int i = 0; i < 8; ++i) a[i] = 0.0f;
  return a;
}

__device__ __forceinline__ v8f wmma16(v16h a, v16h b, v8f c) {
  return __builtin_amdgcn_wmma_f32_16x16x32_f16(false, a, false, b, (short)0, c, false, false);
}

__device__ __forceinline__ float wave_sum(float v) {
  for (int o = 16; o > 0; o >>= 1) v += __shfl_xor(v, o, 32);
  return v;
}
__device__ __forceinline__ float wave_max(float v) {
  for (int o = 16; o > 0; o >>= 1) v = fmaxf(v, __shfl_xor(v, o, 32));
  return v;
}
__device__ __forceinline__ uint32_t pack2(float a, float b) {
  union { _Float16 h[2]; uint32_t u; } t;
  t.h[0] = (_Float16)a; t.h[1] = (_Float16)b;
  return t.u;
}

// ---- prep: f32 -> f16 transposed weights (B^T layout, K contiguous per output column) ----
__global__ void prep_weights(const float* __restrict__ w_kv, const float* __restrict__ w_out,
                             _Float16* __restrict__ ws) {
  const int i = blockIdx.x * 256 + threadIdx.x;
  _Float16* wkvT  = ws;                 // [1024][512]
  _Float16* woutT = ws + 1024 * 512;    // [512][512]
  if (i < 1024 * 512) {
    const int n = i >> 9, k = i & 511;
    wkvT[i] = (_Float16)w_kv[k * 1024 + n];
  } else if (i < 1024 * 512 + 512 * 512) {
    const int j = i - 1024 * 512;
    const int n = j >> 9, k = j & 511;
    woutT[j] = (_Float16)w_out[k * 512 + n];
  }
}

// ---- fused window attention: one workgroup (8 wave32) per 49-token window ----
__global__ __launch_bounds__(256)
void skip_attn_kernel(const float* __restrict__ x, const float* __restrict__ skip,
                      const float* __restrict__ bias_table,
                      const _Float16* __restrict__ wkvT, const _Float16* __restrict__ woutT,
                      float* __restrict__ out) {
  extern __shared__ char smem[];
  _Float16* s_h    = (_Float16*)smem;            // base for all f16 buffers
  _Float16* s_x    = s_h + OFF_X;                // [64][SPAD] skip f16, later reused as O
  _Float16* s_q    = s_h + OFF_Q;                // [64][SPAD] LN(x) * scale
  _Float16* s_k    = s_h + OFF_K;                // [64][SPAD] k raw -> LN in place
  _Float16* s_vT   = s_h + OFF_V;                // [512][VPAD] v transposed
  float*    s_sim  = (float*)(s_vT + DIMX * VPAD);     // [64][64] f32
  _Float16* s_attn = (_Float16*)(s_sim + NPAD * NPAD); // [64][APAD] f16
  float*    s_bias = (float*)(s_attn + NPAD * APAD);   // [169][16] f32

  const int tid = threadIdx.x, lane = tid & 31, wave = tid >> 5;
  const int half = lane >> 4, l16 = lane & 15;
  const int b = blockIdx.x;
  const float* xb = x    + (size_t)b * NTOK * DIMX;
  const float* sb = skip + (size_t)b * NTOK * DIMX;
  float*       ob = out  + (size_t)b * NTOK * DIMX;

  // bias table -> LDS
  for (int i = tid; i < 169 * HEADS; i += 256) s_bias[i] = bias_table[i];

  // zero padded rows 49..63 (cols 0..511) of s_x / s_q
  for (int i = tid; i < (NPAD - NTOK) * (DIMX / 2); i += 256) {
    const int rr = i >> 8, c = i & 255;
    ((uint32_t*)(s_x + (NTOK + rr) * SPAD))[c] = 0u;
    ((uint32_t*)(s_q + (NTOK + rr) * SPAD))[c] = 0u;
  }

  // skip -> f16 into s_x (A operand of kv GEMM)
  for (int i = tid; i < NTOK * (DIMX / 4); i += 256) {
    const int row = i >> 7, c4 = i & 127;
    const float4 v = ((const float4*)(sb + row * DIMX))[c4];
    uint32_t* dst = (uint32_t*)(s_x + row * SPAD);
    dst[c4 * 2 + 0] = pack2(v.x, v.y);
    dst[c4 * 2 + 1] = pack2(v.z, v.w);
  }

  // q = LN(x) * DH^-0.5 -> s_q
  const float qscale = 0.17677669529663689f;
  for (int row = wave; row < NTOK; row += 8) {
    float vals[16];
    float s = 0.f, ss = 0.f;
    const float* xr = xb + row * DIMX;
#pragma unroll
    for (int e = 0; e < 16; ++e) {
      const float v = xr[lane * 16 + e];
      vals[e] = v; s += v; ss += v * v;
    }
    s = wave_sum(s); ss = wave_sum(ss);
    const float mean = s * (1.0f / DIMX);
    const float rstd = rsqrtf(fmaxf(ss * (1.0f / DIMX) - mean * mean, 0.f) + 1e-5f);
#pragma unroll
    for (int e = 0; e < 16; e += 2) {
      ((uint32_t*)(s_q + row * SPAD))[(lane * 16 + e) >> 1] =
          pack2((vals[e] - mean) * rstd * qscale, (vals[e + 1] - mean) * rstd * qscale);
    }
  }
  __syncthreads();

  // ---- stage 1: kv = skip @ w_kv (64x1024, K=512).
  // 4 N-blocks of 256; each wave owns a 64x32 patch (8 acc tiles, 6 frag loads : 8 WMMA per K-step)
#pragma unroll 1
  for (int nb = 0; nb < 2 * DIMX; nb += 256) {
    const int n0 = nb + wave * 32;
    v8f acc[8];
#pragma unroll
    for (int m = 0; m < 8; ++m) acc[m] = zero8();
#pragma unroll 1
    for (int k0 = 0; k0 < DIMX; k0 += 32) {
      const int kp = (k0 + 64) & (DIMX - 1);  // branchless in-bounds prefetch index
      __builtin_prefetch(wkvT + (n0 + l16) * DIMX + kp, 0, 3);
      __builtin_prefetch(wkvT + (n0 + 16 + l16) * DIMX + kp, 0, 3);
      const v16h b0 = load_frag(wkvT, n0,      k0, DIMX, lane);
      const v16h b1 = load_frag(wkvT, n0 + 16, k0, DIMX, lane);
#pragma unroll
      for (int m = 0; m < 4; ++m) {
        const v16h a = load_frag(s_x, m * 16, k0, SPAD, lane);
        acc[m]     = wmma16(a, b0, acc[m]);
        acc[4 + m] = wmma16(a, b1, acc[4 + m]);
      }
    }
    // branchless store-back: select (uniform) offset/stride into the shared LDS block.
    // k   -> s_h[OFF_K + row*SPAD + col]            (row stride SPAD)
    // v^T -> s_h[OFF_V + (col-512)*VPAD + row]      (row stride 1)
    const bool isK = (nb < DIMX);
    const int rstride = isK ? SPAD : 1;
#pragma unroll
    for (int t = 0; t < 2; ++t) {
      const int col = n0 + t * 16 + l16;
      const int base = isK ? (OFF_K + col) : (OFF_V + (col - DIMX) * VPAD);
#pragma unroll
      for (int m = 0; m < 4; ++m)
#pragma unroll
        for (int r = 0; r < 8; ++r) {
          const int row = m * 16 + r + half * 8;
          s_h[base + row * rstride] = (_Float16)acc[t * 4 + m][r];
        }
    }
  }
  __syncthreads();

  // ---- LayerNorm k (in place) and v (via s_vT columns)
#pragma unroll 1
  for (int row = wave; row < NTOK; row += 8) {
    _Float16* kr = s_k + row * SPAD;
    float s = 0.f, ss = 0.f;
#pragma unroll 4
    for (int e = 0; e < 16; ++e) { const float v = (float)kr[lane + e * 32]; s += v; ss += v * v; }
    s = wave_sum(s); ss = wave_sum(ss);
    float mean = s * (1.0f / DIMX);
    float rstd = rsqrtf(fmaxf(ss * (1.0f / DIMX) - mean * mean, 0.f) + 1e-5f);
#pragma unroll 4
    for (int e = 0; e < 16; ++e) {
      const int i = lane + e * 32;
      kr[i] = (_Float16)(((float)kr[i] - mean) * rstd);
    }
    s = 0.f; ss = 0.f;
#pragma unroll 4
    for (int e = 0; e < 16; ++e) {
      const float v = (float)s_vT[(lane + e * 32) * VPAD + row];
      s += v; ss += v * v;
    }
    s = wave_sum(s); ss = wave_sum(ss);
    mean = s * (1.0f / DIMX);
    rstd = rsqrtf(fmaxf(ss * (1.0f / DIMX) - mean * mean, 0.f) + 1e-5f);
#pragma unroll 4
    for (int e = 0; e < 16; ++e) {
      const int i = (lane + e * 32) * VPAD + row;
      s_vT[i] = (_Float16)(((float)s_vT[i] - mean) * rstd);
    }
  }
  __syncthreads();

  // ---- stage 2+3: per-head attention
  const int mt = wave & 3;
#pragma unroll 1
  for (int h = 0; h < HEADS; ++h) {
    const int kq = h * DH;
    // sim = q_h @ k_h^T (+ rel-pos bias, pad mask); 16 tiles / 8 waves = 2 each
#pragma unroll
    for (int t = 0; t < 2; ++t) {
      const int m0 = mt * 16;
      const int n0 = ((wave >> 2) * 2 + t) * 16;
      v8f c = wmma16(load_frag(s_q, m0, kq, SPAD, lane),
                     load_frag(s_k, n0, kq, SPAD, lane), zero8());
      const int col = n0 + l16;
      const int rj = col / 7, cj = col % 7;
#pragma unroll
      for (int r = 0; r < 8; ++r) {
        const int row = m0 + r + half * 8;
        float v = c[r];
        if (col < NTOK) {
          if (row < NTOK) {
            const int ri = row / 7, ci = row % 7;
            v += s_bias[((ri - rj + 6) * 13 + (ci - cj + 6)) * HEADS + h];
          }
        } else {
          v = -1e30f;  // mask padded keys
        }
        s_sim[row * NPAD + col] = v;
      }
    }
    __syncthreads();
    // softmax over 64 cols (8 rows per wave, 2 cols per lane)
#pragma unroll 1
    for (int row = wave; row < NPAD; row += 8) {
      const float v0 = s_sim[row * NPAD + lane];
      const float v1 = s_sim[row * NPAD + lane + 32];
      const float mx = wave_max(fmaxf(v0, v1));
      const float e0 = __expf(v0 - mx), e1 = __expf(v1 - mx);
      const float inv = 1.0f / wave_sum(e0 + e1);
      s_attn[row * APAD + lane]      = (_Float16)(e0 * inv);
      s_attn[row * APAD + lane + 32] = (_Float16)(e1 * inv);
    }
    __syncthreads();
    // O_h = attn @ v_h : 8 tiles (4 m x 2 n), one per wave; K=64 -> 2 steps
    {
      const int m0 = (wave & 3) * 16;
      const int n0 = (wave >> 2) * 16;
      v8f o = zero8();
      o = wmma16(load_frag(s_attn, m0, 0,  APAD, lane),
                 load_frag(s_vT, kq + n0, 0,  VPAD, lane), o);
      o = wmma16(load_frag(s_attn, m0, 32, APAD, lane),
                 load_frag(s_vT, kq + n0, 32, VPAD, lane), o);
      const int col = kq + n0 + l16;
#pragma unroll
      for (int r = 0; r < 8; ++r) {
        const int row = m0 + r + half * 8;
        s_x[row * SPAD + col] = (_Float16)o[r];  // s_x reused as O
      }
    }
    __syncthreads();
  }

  // ---- stage 4: out = O @ w_out (64x512, K=512); each wave: 2 x (64x32 patch)
#pragma unroll 1
  for (int nt = 0; nt < 2; ++nt) {
    const int n0 = wave * 64 + nt * 32;
    v8f acc[8];
#pragma unroll
    for (int m = 0; m < 8; ++m) acc[m] = zero8();
#pragma unroll 1
    for (int k0 = 0; k0 < DIMX; k0 += 32) {
      const int kp = (k0 + 64) & (DIMX - 1);
      __builtin_prefetch(woutT + (n0 + l16) * DIMX + kp, 0, 3);
      __builtin_prefetch(woutT + (n0 + 16 + l16) * DIMX + kp, 0, 3);
      const v16h b0 = load_frag(woutT, n0,      k0, DIMX, lane);
      const v16h b1 = load_frag(woutT, n0 + 16, k0, DIMX, lane);
#pragma unroll
      for (int m = 0; m < 4; ++m) {
        const v16h a = load_frag(s_x, m * 16, k0, SPAD, lane);
        acc[m]     = wmma16(a, b0, acc[m]);
        acc[4 + m] = wmma16(a, b1, acc[4 + m]);
      }
    }
#pragma unroll
    for (int t = 0; t < 2; ++t) {
      const int col = n0 + t * 16 + l16;
#pragma unroll
      for (int m = 0; m < 4; ++m)
#pragma unroll
        for (int r = 0; r < 8; ++r) {
          const int row = m * 16 + r + half * 8;
          if (row < NTOK) ob[row * DIMX + col] = acc[t * 4 + m][r];
        }
    }
  }
}

extern "C" void kernel_launch(void* const* d_in, const int* in_sizes, int n_in,
                              void* d_out, int out_size, void* d_ws, size_t ws_size,
                              hipStream_t stream) {
  (void)in_sizes; (void)n_in; (void)out_size; (void)ws_size;
  const float* x     = (const float*)d_in[0];
  const float* skip  = (const float*)d_in[1];
  const float* w_kv  = (const float*)d_in[2];
  const float* w_out = (const float*)d_in[3];
  const float* bias  = (const float*)d_in[4];

  _Float16* wkvT  = (_Float16*)d_ws;          // 1024x512 f16
  _Float16* woutT = wkvT + 1024 * 512;        // 512x512 f16

  const int total = 1024 * 512 + 512 * 512;
  prep_weights<<<(total + 255) / 256, 256, 0, stream>>>(w_kv, w_out, (_Float16*)d_ws);

  const size_t lds = (size_t)(3 * NPAD * SPAD) * sizeof(_Float16)   // s_x, s_q, s_k
                   + (size_t)(DIMX * VPAD) * sizeof(_Float16)       // s_vT
                   + (size_t)(NPAD * NPAD) * sizeof(float)          // s_sim
                   + (size_t)(NPAD * APAD) * sizeof(_Float16)       // s_attn
                   + (size_t)(169 * HEADS) * sizeof(float);         // s_bias
  (void)hipFuncSetAttribute((const void*)skip_attn_kernel,
                            hipFuncAttributeMaxDynamicSharedMemorySize, (int)lds);
  skip_attn_kernel<<<NWIN, 256, lds, stream>>>(x, skip, bias, wkvT, woutT, (float*)d_out);
}
